// CosineSimilarityModelFERDA_16372415332665
// MI455X (gfx1250) — matrix operand order repaired and checked by reading, not compiled, not run on hardware
//
#include <hip/hip_runtime.h>
#include <hip/hip_bf16.h>

// ---------------- model constants ----------------
#define BT   4        // 2 batches x 2 sequences, batched together
#define SEQ  1024
#define DIM  768
#define NH   12
#define DH   64
#define NL   6
#define FF   3072
#define WIN  128      // one-sided window W2
#define NBLK 8        // SEQ / WIN
#define MTOT (BT*SEQ) // 4096
#define QKVS (3*DIM)  // fused qkv row stride (2304)
#define KOFF (DIM)    // k column offset inside fused qkv
#define VOFF (2*DIM)  // v column offset inside fused qkv

typedef __bf16 bf16;
typedef bf16  bf16x8  __attribute__((ext_vector_type(8)));
typedef bf16  bf16x16 __attribute__((ext_vector_type(16)));
typedef float f32x8   __attribute__((ext_vector_type(8)));

#define BF16CAT(lo,hi) __builtin_shufflevector(lo,hi,0,1,2,3,4,5,6,7,8,9,10,11,12,13,14,15)

static __device__ __forceinline__ f32x8 fzero8(){
  f32x8 z;
#pragma unroll
  for (int j=0;j<8;j++) z[j] = 0.f;
  return z;
}

// ---- CDNA5 async global->LDS copy (ASYNCcnt path), 16B per lane ----
static __device__ __forceinline__ void async_copy_b128(const void* gptr, void* lptr){
  unsigned      laddr = (unsigned)(unsigned long long)lptr;    // LDS offset = addr[31:0]
  unsigned long long gaddr = (unsigned long long)gptr;
  asm volatile("global_load_async_to_lds_b128 %0, %1, off"
               :: "v"(laddr), "v"(gaddr) : "memory");
}
static __device__ __forceinline__ void wait_async(){
  asm volatile("s_wait_asynccnt 0x0" ::: "memory");
}

static __device__ __forceinline__ float blockReduceSum256(float v, float* red){
  int t = threadIdx.x;
  red[t] = v; __syncthreads();
  for (int off = 128; off > 0; off >>= 1){
    if (t < off) red[t] += red[t+off];
    __syncthreads();
  }
  float r = red[0];
  __syncthreads();
  return r;
}

// ---------------- f32 [batch][K][N] -> bf16 [batch][N][K] (weight transpose) ----------------
__global__ void wtrans_kernel(const float* __restrict__ src, bf16* __restrict__ dst,
                              int Kd, int Nd, int batch){
  long total  = (long)batch * Kd * Nd;
  long stride = (long)gridDim.x * blockDim.x;
  for (long idx = (long)blockIdx.x*blockDim.x + threadIdx.x; idx < total; idx += stride){
    long mat = idx / ((long)Kd*Nd);
    long rem = idx - mat*(long)Kd*Nd;
    int  n   = (int)(rem / Kd);
    int  k   = (int)(rem - (long)n*Kd);
    dst[idx] = (bf16)src[mat*(long)Kd*Nd + (long)k*Nd + n];
  }
}

// fused QKV weights: dst[l][3*DIM][DIM] = {Wq^T, Wk^T, Wv^T} per layer
__global__ void wtrans_qkv_kernel(const float* __restrict__ Wq, const float* __restrict__ Wk,
                                  const float* __restrict__ Wv, bf16* __restrict__ dst){
  long total  = (long)NL*3*DIM*DIM;
  long stride = (long)gridDim.x * blockDim.x;
  for (long idx = (long)blockIdx.x*blockDim.x + threadIdx.x; idx < total; idx += stride){
    long l    = idx / (3L*DIM*DIM);
    long rem  = idx - l*3L*DIM*DIM;
    int  sel  = (int)(rem / (DIM*DIM));
    long rem2 = rem - (long)sel*DIM*DIM;
    int  n    = (int)(rem2 / DIM);
    int  k    = (int)(rem2 - (long)n*DIM);
    const float* src = (sel == 0) ? Wq : ((sel == 1) ? Wk : Wv);
    dst[idx] = (bf16)src[l*DIM*DIM + (long)k*DIM + n];
  }
}

// fused QKV bias: dst[l][3*DIM]
__global__ void pack_bias_kernel(const float* __restrict__ bq, const float* __restrict__ bk,
                                 const float* __restrict__ bv, float* __restrict__ dst){
  int idx = blockIdx.x*256 + threadIdx.x;
  if (idx >= NL*3*DIM) return;
  int l = idx / (3*DIM), r = idx % (3*DIM);
  int sel = r / DIM, n = r % DIM;
  const float* s = (sel == 0) ? bq : ((sel == 1) ? bk : bv);
  dst[idx] = s[l*DIM + n];
}

// ---------------- embedding + layernorm ----------------
__global__ __launch_bounds__(256) void embed_ln_kernel(
    const int* __restrict__ x, const float* __restrict__ we, const float* __restrict__ pe,
    const float* __restrict__ te, const float* __restrict__ g, const float* __restrict__ bta,
    float* __restrict__ hF, bf16* __restrict__ hB){
  __shared__ float red[256];
  int tok = blockIdx.x;
  int e = tok / SEQ, s = tok % SEQ;
  int bb = e >> 1, pair = e & 1;
  int id = x[(bb*6 + pair*3 + 0)*SEQ + s];
  int tt = x[(bb*6 + pair*3 + 2)*SEQ + s];
  int t = threadIdx.x;
  float v[3]; float ls = 0.f;
#pragma unroll
  for (int i=0;i<3;i++){
    int d = t + i*256;
    float val = we[(long)id*DIM + d] + pe[(long)(s+2)*DIM + d] + te[(long)tt*DIM + d];
    v[i] = val; ls += val;
  }
  float mean = blockReduceSum256(ls, red) * (1.0f/DIM);
  float lv = 0.f;
#pragma unroll
  for (int i=0;i<3;i++){ float d0 = v[i]-mean; lv += d0*d0; }
  float var  = blockReduceSum256(lv, red) * (1.0f/DIM);
  float rinv = rsqrtf(var + 1e-5f);
#pragma unroll
  for (int i=0;i<3;i++){
    int d = t + i*256;
    float o = (v[i]-mean)*rinv*g[d] + bta[d];
    hF[(long)tok*DIM + d] = o;
    hB[(long)tok*DIM + d] = (bf16)o;
  }
}

// ---------------- residual add + layernorm ----------------
__global__ __launch_bounds__(256) void add_ln_kernel(
    const float* __restrict__ base, const float* __restrict__ delta,
    const float* __restrict__ g, const float* __restrict__ bta,
    float* __restrict__ hF, bf16* __restrict__ hB){
  __shared__ float red[256];
  long tok = blockIdx.x;
  int  t   = threadIdx.x;
  float v[3]; float ls = 0.f;
#pragma unroll
  for (int i=0;i<3;i++){
    int d = t + i*256;
    float val = base[tok*DIM + d] + delta[tok*DIM + d];
    v[i] = val; ls += val;
  }
  float mean = blockReduceSum256(ls, red) * (1.0f/DIM);
  float lv = 0.f;
#pragma unroll
  for (int i=0;i<3;i++){ float d0 = v[i]-mean; lv += d0*d0; }
  float var  = blockReduceSum256(lv, red) * (1.0f/DIM);
  float rinv = rsqrtf(var + 1e-5f);
#pragma unroll
  for (int i=0;i<3;i++){
    int d = t + i*256;
    float o = (v[i]-mean)*rinv*g[d] + bta[d];
    hF[tok*DIM + d] = o;
    hB[tok*DIM + d] = (bf16)o;
  }
}

// ---------------- WMMA GEMM: C[.,N] = op(A[.,K] x Wt[N,K]^T + bias) ----------------
// Double-buffered LDS, async global->LDS staging, 8 waves as 4(M) x 2(N), 32x32 per wave.
// Steady-state loop always prefetches (last iteration peeled -> branch-free inner loop).
// op: 0 = f32 out, 1 = bf16 out, 2 = gelu(exact)+bf16 out
static __device__ __forceinline__ void gemm_compute_step(
    const bf16* __restrict__ lA, const bf16* __restrict__ lB,
    int mw, int nw, int half, int l16, f32x8 acc[2][2]){
#pragma unroll
  for (int sm=0; sm<2; sm++){
    int row = mw*32 + sm*16 + l16;
    bf16x8 alo = *(const bf16x8*)&lA[row*32 + half*8];
    bf16x8 ahi = *(const bf16x8*)&lA[row*32 + 16 + half*8];
    bf16x16 afrag = BF16CAT(alo, ahi);
#pragma unroll
    for (int sn=0; sn<2; sn++){
      int col = nw*32 + sn*16 + l16;
      bf16x8 blo = *(const bf16x8*)&lB[col*32 + half*8];
      bf16x8 bhi = *(const bf16x8*)&lB[col*32 + 16 + half*8];
      bf16x16 bfrag = BF16CAT(blo, bhi);
      acc[sm][sn] = __builtin_amdgcn_wmma_f32_16x16x32_bf16(
          false, afrag, false, bfrag, (short)0, acc[sm][sn], false, false);
    }
  }
}

__global__ __launch_bounds__(256) void gemm_bf16_kernel(
    const bf16* __restrict__ A, const bf16* __restrict__ Wt, const float* __restrict__ bias,
    float* __restrict__ outF, bf16* __restrict__ outB, int N, int K, int op){
  __shared__ bf16 ldsA[2][128*32];   // 2 x 8KB
  __shared__ bf16 ldsB[2][64*32];    // 2 x 4KB
  int t    = threadIdx.x;
  int wave = t >> 5, lane = t & 31, half = lane >> 4, l16 = lane & 15;
  int mw   = wave & 3, nw = wave >> 2;      // 4 x 2 wave grid, 32x32 tile per wave
  int bn   = blockIdx.x;
  long arow = (long)blockIdx.y * 128;

  // per-thread staging geometry (fixed across K-steps)
  int cc0 = t*2, cc1 = t*2 + 1;                   // A: 512 chunks of 16B
  int rA0 = cc0 >> 2, kA0 = (cc0 & 3)*8;
  int rA1 = cc1 >> 2, kA1 = (cc1 & 3)*8;
  int rB  = t >> 2,   kB  = (t & 3)*8;            // B: 256 chunks of 16B
  const bf16* gA0 = &A [(arow + rA0)*K + kA0];
  const bf16* gA1 = &A [(arow + rA1)*K + kA1];
  const bf16* gB  = &Wt[((long)bn*64 + rB)*K + kB];
  int  oA0 = rA0*32 + kA0, oA1 = rA1*32 + kA1, oB = rB*32 + kB;

  f32x8 acc[2][2];
#pragma unroll
  for (int i=0;i<2;i++)
#pragma unroll
    for (int j=0;j<2;j++) acc[i][j] = fzero8();

  int nt = K >> 5;                  // always >= 2 here (K = 768 or 3072)
  // prologue: stage tile 0 into buffer 0
  async_copy_b128(gA0, &ldsA[0][oA0]);
  async_copy_b128(gA1, &ldsA[0][oA1]);
  async_copy_b128(gB,  &ldsB[0][oB]);
  wait_async();
  __syncthreads();

  int buf = 0;
  for (int i = 0; i < nt - 1; ++i){          // branch-free steady state: always prefetch
    int k1 = (i + 1) << 5;
    async_copy_b128(gA0 + k1, &ldsA[buf^1][oA0]);
    async_copy_b128(gA1 + k1, &ldsA[buf^1][oA1]);
    async_copy_b128(gB  + k1, &ldsB[buf^1][oB]);
    gemm_compute_step(&ldsA[buf][0], &ldsB[buf][0], mw, nw, half, l16, acc);
    wait_async();
    __syncthreads();
    buf ^= 1;
  }
  gemm_compute_step(&ldsA[buf][0], &ldsB[buf][0], mw, nw, half, l16, acc);  // last tile

#pragma unroll
  for (int sn=0;sn<2;sn++){
    int col  = bn*64 + nw*32 + sn*16 + l16;
    float bv = bias[col];
#pragma unroll
    for (int sm=0;sm<2;sm++){
#pragma unroll
      for (int r=0;r<8;r++){
        long row = arow + mw*32 + sm*16 + r + half*8;
        float v  = acc[sm][sn][r] + bv;
        if (op == 2) v = 0.5f*v*(1.0f + erff(v*0.70710678118f));
        if (op == 0) outF[row*N + col] = v;
        else         outB[row*N + col] = (bf16)v;
      }
    }
  }
}

// ---------------- sliding-window scores (+1/8 scale) + masks + fused softmax -> bf16 probs ----
__global__ __launch_bounds__(256) void attn_scores_kernel(
    const bf16* __restrict__ QKV, const int* __restrict__ x, bf16* __restrict__ P){
  __shared__ bf16 ldsK[3*WIN*DH];          // [kk][d], 48KB
  int blk = blockIdx.x;                    // bt*NH*NBLK + h*NBLK + nb
  int bt  = blk / (NH*NBLK);
  int rem = blk % (NH*NBLK);
  int h   = rem / NBLK, nb = rem % NBLK;
  int t = threadIdx.x, wave = t>>5, lane = t&31, half = lane>>4, l16 = lane&15;

  // stage K3 via async DMA; zero-fill out-of-range rows with plain LDS stores
#pragma unroll
  for (int i=0;i<12;i++){
    int c   = t + i*256;                   // 384 rows x 8 chunks
    int row = c >> 3, coff = (c & 7)*8;
    int g   = nb*WIN - WIN + row;
    if (g >= 0 && g < SEQ){
      async_copy_b128(&QKV[((long)bt*SEQ + g)*QKVS + KOFF + h*DH + coff],
                      &ldsK[row*DH + coff]);
    } else {
      uint4 z; z.x = z.y = z.z = z.w = 0u;
      *(uint4*)&ldsK[row*DH + coff] = z;
    }
  }
  wait_async();
  __syncthreads();

  f32x8 acc[24];
#pragma unroll
  for (int i=0;i<24;i++) acc[i] = fzero8();

  const bf16* qrow = QKV + ((long)bt*SEQ + nb*WIN + wave*16 + l16)*QKVS + h*DH;
#pragma unroll
  for (int k0=0;k0<DH;k0+=32){
    bf16x8 alo = *(const bf16x8*)&qrow[k0 + half*8];
    bf16x8 ahi = *(const bf16x8*)&qrow[k0 + 16 + half*8];
    bf16x16 afrag = BF16CAT(alo, ahi);
#pragma unroll
    for (int tn=0;tn<24;tn++){
      bf16x8 blo = *(const bf16x8*)&ldsK[(tn*16 + l16)*DH + k0 + half*8];
      bf16x8 bhi = *(const bf16x8*)&ldsK[(tn*16 + l16)*DH + k0 + 16 + half*8];
      bf16x16 bfrag = BF16CAT(blo, bhi);
      acc[tn] = __builtin_amdgcn_wmma_f32_16x16x32_bf16(
          false, afrag, false, bfrag, (short)0, acc[tn], false, false);
    }
  }

  const int* mrow = x + ((bt>>1)*6 + (bt&1)*3 + 1)*SEQ;
  float km[24];
#pragma unroll
  for (int tn=0;tn<24;tn++){
    int n = tn*16 + l16;
    int g = nb*WIN - WIN + n;
    km[tn] = (g >= 0 && g < SEQ && mrow[g] != 0) ? 0.f : -1e9f;
  }

  bf16* prow = P + (long)blk*(WIN*3*WIN);
#pragma unroll
  for (int r=0;r<8;r++){
    int qr = wave*16 + r + half*8;
    float mx = -3.0e38f;
#pragma unroll
    for (int tn=0;tn<24;tn++){
      int n   = tn*16 + l16;
      int rel = n - WIN - qr;
      float v = acc[tn][r]*0.125f + km[tn] + ((rel >= -WIN && rel <= WIN) ? 0.f : -1e9f);
      acc[tn][r] = v;
      mx = fmaxf(mx, v);
    }
#pragma unroll
    for (int m=1;m<16;m<<=1) mx = fmaxf(mx, __shfl_xor(mx, m));
    float sum = 0.f;
#pragma unroll
    for (int tn=0;tn<24;tn++){
      float e = __expf(acc[tn][r] - mx);
      acc[tn][r] = e; sum += e;
    }
#pragma unroll
    for (int m=1;m<16;m<<=1) sum += __shfl_xor(sum, m);
    float inv = 1.0f / sum;
#pragma unroll
    for (int tn=0;tn<24;tn++){
      int n = tn*16 + l16;
      prow[(long)qr*(3*WIN) + n] = (bf16)(acc[tn][r]*inv);
    }
  }
}

// ---------------- probs @ V3 -> attention output (bf16, head-interleaved) ----------------
__global__ __launch_bounds__(256) void attn_pv_kernel(
    const bf16* __restrict__ P, const bf16* __restrict__ QKV, bf16* __restrict__ Out){
  __shared__ bf16 ldsVT[DH*3*WIN];         // [d][kk], 48KB
  int blk = blockIdx.x;
  int bt  = blk / (NH*NBLK);
  int rem = blk % (NH*NBLK);
  int h   = rem / NBLK, nb = rem % NBLK;
  int t = threadIdx.x, wave = t>>5, lane = t&31, half = lane>>4, l16 = lane&15;

  // batched register staging, then transposed scatter into LDS
  bf16x8 vreg[12];
#pragma unroll
  for (int i=0;i<12;i++){
    int c   = t + i*256;
    int row = c >> 3, coff = (c & 7)*8;
    int g   = nb*WIN - WIN + row;
    bf16x8 v;
#pragma unroll
    for (int j=0;j<8;j++) v[j] = (bf16)0.f;
    if (g >= 0 && g < SEQ)
      v = *(const bf16x8*)&QKV[((long)bt*SEQ + g)*QKVS + VOFF + h*DH + coff];
    vreg[i] = v;
  }
#pragma unroll
  for (int i=0;i<12;i++){
    int c   = t + i*256;
    int row = c >> 3, coff = (c & 7)*8;
#pragma unroll
    for (int j=0;j<8;j++) ldsVT[(coff + j)*(3*WIN) + row] = vreg[i][j];
  }
  __syncthreads();

  f32x8 acc[4];
#pragma unroll
  for (int i=0;i<4;i++) acc[i] = fzero8();

  const bf16* prow = P + (long)blk*(WIN*3*WIN) + (long)(wave*16 + l16)*(3*WIN);
  for (int k0=0;k0<3*WIN;k0+=32){
    bf16x8 alo = *(const bf16x8*)&prow[k0 + half*8];
    bf16x8 ahi = *(const bf16x8*)&prow[k0 + 16 + half*8];
    bf16x16 afrag = BF16CAT(alo, ahi);
#pragma unroll
    for (int tn=0;tn<4;tn++){
      bf16x8 blo = *(const bf16x8*)&ldsVT[(tn*16 + l16)*(3*WIN) + k0 + half*8];
      bf16x8 bhi = *(const bf16x8*)&ldsVT[(tn*16 + l16)*(3*WIN) + k0 + 16 + half*8];
      bf16x16 bfrag = BF16CAT(blo, bhi);
      acc[tn] = __builtin_amdgcn_wmma_f32_16x16x32_bf16(
          false, afrag, false, bfrag, (short)0, acc[tn], false, false);
    }
  }

#pragma unroll
  for (int tn=0;tn<4;tn++){
    int d = tn*16 + l16;
#pragma unroll
    for (int r=0;r<8;r++){
      int qr = wave*16 + r + half*8;
      Out[((long)bt*SEQ + nb*WIN + qr)*DIM + h*DH + d] = (bf16)acc[tn][r];
    }
  }
}

// ---------------- pooler: tanh(h[:,0] @ Wp + bp) ----------------
__global__ __launch_bounds__(256) void pooler_kernel(
    const float* __restrict__ hF, const float* __restrict__ Wp, const float* __restrict__ bp,
    float* __restrict__ pooled){
  int idx = blockIdx.x*256 + threadIdx.x;
  int e = idx / DIM, n = idx % DIM;
  const float* hr = hF + (long)e*SEQ*DIM;
  float acc = bp[n];
  for (int d=0; d<DIM; d++) acc += hr[d]*Wp[(long)d*DIM + n];
  pooled[(long)e*DIM + n] = tanhf(acc);
}

// ---------------- cosine similarity head ----------------
__global__ __launch_bounds__(256) void final_kernel(const float* __restrict__ pooled,
                                                    float* __restrict__ out){
  __shared__ float red[256];
  int t = threadIdx.x;
  for (int b=0;b<2;b++){
    const float* p1 = pooled + (long)(2*b  )*DIM;
    const float* p2 = pooled + (long)(2*b+1)*DIM;
    float d=0.f, n1=0.f, n2=0.f;
    for (int i=t;i<DIM;i+=256){ float a=p1[i], c=p2[i]; d+=a*c; n1+=a*a; n2+=c*c; }
    d  = blockReduceSum256(d,  red);
    n1 = blockReduceSum256(n1, red);
    n2 = blockReduceSum256(n2, red);
    if (t == 0){
      float denom = fmaxf(sqrtf(n1)*sqrtf(n2), 1e-8f);
      float pred  = (d/denom + 1.f)*0.5f;
      out[b*2+0] = 1.f - pred;
      out[b*2+1] = pred;
    }
    __syncthreads();
  }
}

// ---------------- host orchestration ----------------
extern "C" void kernel_launch(void* const* d_in, const int* in_sizes, int n_in,
                              void* d_out, int out_size, void* d_ws, size_t ws_size,
                              hipStream_t stream){
  (void)in_sizes; (void)n_in; (void)out_size; (void)ws_size;
  const int*   x        = (const int*)  d_in[0];
  const float* word_emb = (const float*)d_in[1];
  const float* pos_emb  = (const float*)d_in[2];
  const float* type_emb = (const float*)d_in[3];
  const float* ln_e_g   = (const float*)d_in[4];
  const float* ln_e_b   = (const float*)d_in[5];
  const float* Wq = (const float*)d_in[6];   const float* bq = (const float*)d_in[7];
  const float* Wk = (const float*)d_in[8];   const float* bk = (const float*)d_in[9];
  const float* Wv = (const float*)d_in[10];  const float* bv = (const float*)d_in[11];
  const float* Wo = (const float*)d_in[12];  const float* bo = (const float*)d_in[13];
  const float* ln1_g = (const float*)d_in[14]; const float* ln1_b = (const float*)d_in[15];
  const float* W1 = (const float*)d_in[16];  const float* b1 = (const float*)d_in[17];
  const float* W2m = (const float*)d_in[18]; const float* b2 = (const float*)d_in[19];
  const float* ln2_g = (const float*)d_in[20]; const float* ln2_b = (const float*)d_in[21];
  const float* Wp = (const float*)d_in[22];  const float* bp = (const float*)d_in[23];
  float* out = (float*)d_out;

  char* ws = (char*)d_ws;
  size_t off = 0;
  auto alloc = [&](size_t nbytes)->char*{
    char* p = ws + off;
    off += (nbytes + 255) & ~(size_t)255;
    return p;
  };
  bf16*  qkvT  = (bf16*) alloc((size_t)NL*3*DIM*DIM*2);   // [l][3*DIM][DIM]
  bf16*  woT   = (bf16*) alloc((size_t)NL*DIM*DIM*2);
  bf16*  w1T   = (bf16*) alloc((size_t)NL*FF*DIM*2);      // [l][FF][DIM]
  bf16*  w2T   = (bf16*) alloc((size_t)NL*DIM*FF*2);      // [l][DIM][FF]
  float* bqkv  = (float*)alloc((size_t)NL*3*DIM*4);       // [l][3*DIM]
  float* hF    = (float*)alloc((size_t)MTOT*DIM*4);
  bf16*  hB    = (bf16*) alloc((size_t)MTOT*DIM*2);
  bf16*  qkvB  = (bf16*) alloc((size_t)MTOT*QKVS*2);      // [tok][3*DIM]
  bf16*  aoB   = (bf16*) alloc((size_t)MTOT*DIM*2);
  float* projF = (float*)alloc((size_t)MTOT*DIM*4);
  bf16*  ffn1B = (bf16*) alloc((size_t)MTOT*FF*2);
  bf16*  probsB= (bf16*) alloc((size_t)BT*NH*NBLK*WIN*3*WIN*2);
  float* pooled= (float*)alloc((size_t)BT*DIM*4);

  // weight conversion (f32 -> bf16, transposed to [N][K])
  wtrans_qkv_kernel<<<4096,256,0,stream>>>(Wq, Wk, Wv, qkvT);
  wtrans_kernel<<<2048,256,0,stream>>>(Wo,  woT, DIM, DIM, NL);
  wtrans_kernel<<<4096,256,0,stream>>>(W1,  w1T, DIM, FF,  NL);
  wtrans_kernel<<<4096,256,0,stream>>>(W2m, w2T, FF,  DIM, NL);
  pack_bias_kernel<<<(NL*3*DIM+255)/256,256,0,stream>>>(bq, bk, bv, bqkv);

  embed_ln_kernel<<<MTOT,256,0,stream>>>(x, word_emb, pos_emb, type_emb, ln_e_g, ln_e_b, hF, hB);

  dim3 gQKV(QKVS/64, MTOT/128);  // 36 x 32
  dim3 gD(DIM/64,    MTOT/128);  // 12 x 32
  dim3 gF(FF/64,     MTOT/128);  // 48 x 32
  int  attnBlocks = BT*NH*NBLK;  // 384

  for (int l=0; l<NL; l++){
    const bf16* qkv_l = qkvT + (size_t)l*3*DIM*DIM;
    const bf16* wo_l  = woT  + (size_t)l*DIM*DIM;
    const bf16* w1_l  = w1T  + (size_t)l*FF*DIM;
    const bf16* w2_l  = w2T  + (size_t)l*DIM*FF;

    gemm_bf16_kernel<<<gQKV,256,0,stream>>>(hB, qkv_l, bqkv + l*3*DIM, nullptr, qkvB, QKVS, DIM, 1);

    attn_scores_kernel<<<attnBlocks,256,0,stream>>>(qkvB, x, probsB);
    attn_pv_kernel<<<attnBlocks,256,0,stream>>>(probsB, qkvB, aoB);

    gemm_bf16_kernel<<<gD,256,0,stream>>>(aoB, wo_l, bo + l*DIM, projF, nullptr, DIM, DIM, 0);
    add_ln_kernel<<<MTOT,256,0,stream>>>(hF, projF, ln1_g + l*DIM, ln1_b + l*DIM, hF, hB);

    gemm_bf16_kernel<<<gF,256,0,stream>>>(hB, w1_l, b1 + l*FF, nullptr, ffn1B, FF, DIM, 2);
    gemm_bf16_kernel<<<gD,256,0,stream>>>(ffn1B, w2_l, b2 + l*DIM, projF, nullptr, DIM, FF, 0);
    add_ln_kernel<<<MTOT,256,0,stream>>>(hF, projF, ln2_g + l*DIM, ln2_b + l*DIM, hF, hB);
  }

  pooler_kernel<<<(BT*DIM)/256,256,0,stream>>>(hF, Wp, bp, pooled);
  final_kernel<<<1,256,0,stream>>>(pooled, out);
}
